// BasisVQ_11845519802661
// MI455X (gfx1250) — matrix operand order repaired and checked by reading, not compiled, not run on hardware
//
#include <hip/hip_runtime.h>
#include <stdint.h>

#define BKROWS   32768            // B*K = 128*256 rows
#define DM       256              // d_model (K of GEMM1)
#define NC       1024             // num codes
#define BD       2700             // basis dim
#define BD_PAD   2720             // padded to multiple of 32 (85 K-steps)
#define NPROJ_PAD 2816            // W rows padded to 11*256
#define QSIZE    ((size_t)BKROWS * (size_t)BD)

typedef __attribute__((ext_vector_type(16))) __bf16 bf16x16;
typedef __attribute__((ext_vector_type(8)))  float  floatx8;

__device__ inline bf16x16 make_frag(const __bf16* lo, const __bf16* hi) {
    uint4 t[2];
    t[0] = *reinterpret_cast<const uint4*>(lo);
    t[1] = *reinterpret_cast<const uint4*>(hi);
    bf16x16 r;
    __builtin_memcpy(&r, t, sizeof(r));
    return r;
}

// ---------------- prep kernels ----------------

__global__ void cvt_f32_bf16(const float* __restrict__ in, __bf16* __restrict__ out, int n) {
    int i = blockIdx.x * 256 + threadIdx.x;
    if (i < n) out[i] = (__bf16)in[i];
}

// W_proj [2700,256] -> bf16 padded to [2816,256], pad rows zero
__global__ void cvt_w_pad(const float* __restrict__ W, __bf16* __restrict__ out) {
    int i = blockIdx.x * 256 + threadIdx.x;   // over NPROJ_PAD*DM
    int row = i / DM;
    float v = (row < BD) ? W[i] : 0.0f;
    out[i] = (__bf16)v;
}

// basis [1024,2700] -> bf16 padded to [1024,2720] + row norms (f32, from f32 data)
__global__ void cvt_basis(const float* __restrict__ basis, __bf16* __restrict__ out,
                          float* __restrict__ bnorm) {
    __shared__ float red[256];
    int row = blockIdx.x, t = threadIdx.x;
    float acc = 0.0f;
    for (int j = t; j < BD_PAD; j += 256) {
        float v = (j < BD) ? basis[(size_t)row * BD + j] : 0.0f;
        out[(size_t)row * BD_PAD + j] = (__bf16)v;
        acc += v * v;
    }
    red[t] = acc; __syncthreads();
    for (int s = 128; s > 0; s >>= 1) { if (t < s) red[t] += red[t + s]; __syncthreads(); }
    if (t == 0) bnorm[row] = red[0];
}

__global__ void zero_accums(float* __restrict__ avg, float* __restrict__ loss) {
    int i = blockIdx.x * 256 + threadIdx.x;
    if (i < NC) avg[i] = 0.0f;
    if (i == NC) loss[0] = 0.0f;
}

// ---------------- WMMA GEMM: C[M,N] = A[M,K] * B[N,K]^T ----------------
// Block tile 64(M) x 256(N), 8 waves (2x4), each wave 32x64: 2 A-frags x 4 B-frags
// = 8 WMMAs per K-step. Double-buffered LDS (stride-40 rows, conflict-free b128),
// global loads for tile s+1 staged in registers while tile s is consumed by WMMA.
__global__ __launch_bounds__(256)
void wmma_gemm_bf16(const __bf16* __restrict__ A, const __bf16* __restrict__ Bm,
                    int lda, int ldb, int ksteps,
                    float*  __restrict__ Cf32, int ldc,                  // optional f32 out
                    __bf16* __restrict__ Cbf,  int ldcb,                 // optional bf16 out
                    const float* __restrict__ bias, int nvalid, int nstore_max) {
    __shared__ __bf16 ldsA[2][64 * 40];
    __shared__ __bf16 ldsB[2][256 * 40];

    const int t    = threadIdx.x;
    const int m0   = blockIdx.x * 64;
    const int n0   = blockIdx.y * 256;
    const int w    = t >> 5;
    const int lane = t & 31;
    const int half = (lane < 16) ? 0 : 1;
    const int l16  = lane & 15;
    const int m_off = (w & 1) * 32;     // waves: 2 in M
    const int n_off = (w >> 1) * 64;    // 4 in N

    floatx8 acc[2][4];
#pragma unroll
    for (int i = 0; i < 2; ++i)
#pragma unroll
        for (int j = 0; j < 4; ++j) acc[i][j] = (floatx8){};

    const int arow = t >> 2;              // 0..63
    const int ch   = (t & 3) * 8;         // 0,8,16,24
    const __bf16* Aptr  = A  + (size_t)(m0 + arow) * lda + ch;
    const __bf16* Bptr  = Bm + (size_t)(n0 + arow) * ldb + ch;
    const size_t  bstep = (size_t)64 * ldb;

    uint4 ra, rb0, rb1, rb2, rb3;
    // preload K-tile 0 straight into buffer 0
    ra  = *reinterpret_cast<const uint4*>(Aptr);
    rb0 = *reinterpret_cast<const uint4*>(Bptr);
    rb1 = *reinterpret_cast<const uint4*>(Bptr + bstep);
    rb2 = *reinterpret_cast<const uint4*>(Bptr + 2 * bstep);
    rb3 = *reinterpret_cast<const uint4*>(Bptr + 3 * bstep);
    *reinterpret_cast<uint4*>(&ldsA[0][arow * 40 + ch])         = ra;
    *reinterpret_cast<uint4*>(&ldsB[0][arow * 40 + ch])         = rb0;
    *reinterpret_cast<uint4*>(&ldsB[0][(arow + 64) * 40 + ch])  = rb1;
    *reinterpret_cast<uint4*>(&ldsB[0][(arow + 128) * 40 + ch]) = rb2;
    *reinterpret_cast<uint4*>(&ldsB[0][(arow + 192) * 40 + ch]) = rb3;

    for (int s = 0; s < ksteps; ++s) {
        const int cur = s & 1;
        const int nxt = cur ^ 1;
        __syncthreads();                      // buf[cur] is ready for all waves

        const bool more = (s + 1) < ksteps;   // wave-uniform
        if (more) {                           // stage next global tile in regs
            const int kk = (s + 1) * 32;
            ra  = *reinterpret_cast<const uint4*>(Aptr + kk);
            rb0 = *reinterpret_cast<const uint4*>(Bptr + kk);
            rb1 = *reinterpret_cast<const uint4*>(Bptr + bstep + kk);
            rb2 = *reinterpret_cast<const uint4*>(Bptr + 2 * bstep + kk);
            rb3 = *reinterpret_cast<const uint4*>(Bptr + 3 * bstep + kk);
        }

        // A fragments: lane half 0 -> K {0..7,16..23}; half 1 -> K {8..15,24..31}
        const int aRow = m_off + l16;
        bf16x16 af0 = make_frag(&ldsA[cur][aRow * 40 + half * 8],
                                &ldsA[cur][aRow * 40 + half * 8 + 16]);
        bf16x16 af1 = make_frag(&ldsA[cur][(aRow + 16) * 40 + half * 8],
                                &ldsA[cur][(aRow + 16) * 40 + half * 8 + 16]);
#pragma unroll
        for (int j = 0; j < 4; ++j) {
            // B fragment: N = lane%16; half 0 -> K {0..7,16..23}; half 1 -> K {8..15,24..31}
            const int bRow = n_off + j * 16 + l16;
            bf16x16 bf = make_frag(&ldsB[cur][bRow * 40 + half * 8],
                                   &ldsB[cur][bRow * 40 + half * 8 + 16]);
            acc[0][j] = __builtin_amdgcn_wmma_f32_16x16x32_bf16(
                false, af0, false, bf, (short)0, acc[0][j], false, false);
            acc[1][j] = __builtin_amdgcn_wmma_f32_16x16x32_bf16(
                false, af1, false, bf, (short)0, acc[1][j], false, false);
        }

        if (more) {                           // commit staged regs to other buffer
            *reinterpret_cast<uint4*>(&ldsA[nxt][arow * 40 + ch])         = ra;
            *reinterpret_cast<uint4*>(&ldsB[nxt][arow * 40 + ch])         = rb0;
            *reinterpret_cast<uint4*>(&ldsB[nxt][(arow + 64) * 40 + ch])  = rb1;
            *reinterpret_cast<uint4*>(&ldsB[nxt][(arow + 128) * 40 + ch]) = rb2;
            *reinterpret_cast<uint4*>(&ldsB[nxt][(arow + 192) * 40 + ch]) = rb3;
        }
    }

    // epilogue: D layout — VGPR v, lane half 0 -> M=v, half 1 -> M=v+8; N = lane%16
#pragma unroll
    for (int i = 0; i < 2; ++i) {
#pragma unroll
        for (int j = 0; j < 4; ++j) {
            const int nG = n0 + n_off + j * 16 + l16;
            float bv = 0.0f;
            if (bias != nullptr && nG < nvalid) bv = bias[nG];
#pragma unroll
            for (int v = 0; v < 8; ++v) {
                const int row = m0 + m_off + i * 16 + v + half * 8;
                const float val = acc[i][j][v] + bv;
                if (Cf32 != nullptr) Cf32[(size_t)row * ldc + nG] = val;
                if (Cbf != nullptr && nG < nstore_max)
                    Cbf[(size_t)row * ldcb + nG] = (nG < nvalid) ? (__bf16)val : (__bf16)0.0f;
            }
        }
    }
}

// ---------------- argmin + softmax accumulation ----------------
// s_c = ||b_c||^2 - 2*dot  (per-row ||z||^2 cancels in both argmin and softmax)
__global__ __launch_bounds__(256)
void vq_reduce(const float* __restrict__ score, const float* __restrict__ bnorm,
               float* __restrict__ avg, int* __restrict__ idx_out,
               float* __restrict__ idx_f) {
    __shared__ float probs[NC];
    __shared__ float bn_s[NC];
    __shared__ float redv[256];
    __shared__ int   redi[256];
    __shared__ float rmin_s, rsum_s;
    __shared__ int   ridx_s;
    const int t = threadIdx.x;
#pragma unroll
    for (int i = 0; i < 4; ++i) { probs[t + i * 256] = 0.0f; bn_s[t + i * 256] = bnorm[t + i * 256]; }
    __syncthreads();

    for (int r = 0; r < 8; ++r) {
        const int row = blockIdx.x * 8 + r;
        const float* sr = score + (size_t)row * NC;
        float sv[4]; float mn = 3.4e38f; int mi = 0;
#pragma unroll
        for (int i = 0; i < 4; ++i) {
            const int c = t + i * 256;
            sv[i] = bn_s[c] - 2.0f * sr[c];
            if (sv[i] < mn) { mn = sv[i]; mi = c; }
        }
        redv[t] = mn; redi[t] = mi; __syncthreads();
        for (int s2 = 128; s2 > 0; s2 >>= 1) {
            if (t < s2) {
                const float ov = redv[t + s2]; const int oi = redi[t + s2];
                if (ov < redv[t] || (ov == redv[t] && oi < redi[t])) { redv[t] = ov; redi[t] = oi; }
            }
            __syncthreads();
        }
        if (t == 0) { rmin_s = redv[0]; ridx_s = redi[0]; }
        __syncthreads();

        float ev[4]; float ps = 0.0f;
#pragma unroll
        for (int i = 0; i < 4; ++i) { ev[i] = expf(rmin_s - sv[i]); ps += ev[i]; }
        redv[t] = ps; __syncthreads();
        for (int s2 = 128; s2 > 0; s2 >>= 1) { if (t < s2) redv[t] += redv[t + s2]; __syncthreads(); }
        if (t == 0) rsum_s = redv[0];
        __syncthreads();

        const float inv = 1.0f / rsum_s;
#pragma unroll
        for (int i = 0; i < 4; ++i) probs[t + i * 256] += ev[i] * inv;
        if (t == 0) { idx_out[row] = ridx_s; idx_f[row] = (float)ridx_s; }
        __syncthreads();
    }
#pragma unroll
    for (int i = 0; i < 4; ++i) {
        const int c = t + i * 256;
        atomicAdd(&avg[c], probs[c] * (1.0f / 32768.0f));
    }
}

// ---------------- q_st gather + commit loss ----------------
__global__ __launch_bounds__(256)
void vq_gather(const float* __restrict__ basis, const __bf16* __restrict__ zbf,
               const int* __restrict__ idx, float* __restrict__ q,
               float* __restrict__ loss) {
    __shared__ float red[256];
    const int row = blockIdx.x, t = threadIdx.x;
    const int ci = idx[row];
    const float* br = basis + (size_t)ci * BD;
    const __bf16* zr = zbf + (size_t)row * BD_PAD;
    float acc = 0.0f;
    for (int j = t; j < BD; j += 256) {
        const float e = br[j];
        q[(size_t)row * BD + j] = e;
        const float z = (float)zr[j];
        const float d = z - e;
        acc += d * d;
    }
    red[t] = acc; __syncthreads();
    for (int s = 128; s > 0; s >>= 1) { if (t < s) red[t] += red[t + s]; __syncthreads(); }
    if (t == 0) atomicAdd(loss, red[0]);
}

__global__ void vq_final(const float* __restrict__ avg, const float* __restrict__ loss,
                         float* __restrict__ out_scalars) {
    __shared__ float red[256];
    const int t = threadIdx.x;
    float s = 0.0f;
#pragma unroll
    for (int i = 0; i < 4; ++i) { const float p = avg[t + i * 256]; s += p * logf(p + 1e-8f); }
    red[t] = s; __syncthreads();
    for (int st = 128; st > 0; st >>= 1) { if (t < st) red[t] += red[t + st]; __syncthreads(); }
    if (t == 0) {
        out_scalars[0] = 0.25f * loss[0] / ((float)BKROWS * (float)BD);  // vq_loss
        out_scalars[1] = -red[0];                                        // entropy
    }
}

// ---------------- launcher ----------------
extern "C" void kernel_launch(void* const* d_in, const int* in_sizes, int n_in,
                              void* d_out, int out_size, void* d_ws, size_t ws_size,
                              hipStream_t stream) {
    const float* X     = (const float*)d_in[0];   // [32768,256]
    const float* W     = (const float*)d_in[1];   // [2700,256]
    const float* bprj  = (const float*)d_in[2];   // [2700]
    const float* basis = (const float*)d_in[3];   // [1024,2700]

    float* out   = (float*)d_out;
    float* q     = out;                 // [32768,2700]
    float* idx_f = out + QSIZE;         // [32768]
    float* scal  = idx_f + BKROWS;      // vq_loss, entropy

    char* ws = (char*)d_ws;
    size_t off = 0;
    auto alloc = [&](size_t bytes) -> char* {
        char* p = ws + off;
        off = (off + bytes + 255) & ~(size_t)255;
        return p;
    };
    __bf16* Xb    = (__bf16*)alloc((size_t)BKROWS * DM * 2);
    __bf16* Wb    = (__bf16*)alloc((size_t)NPROJ_PAD * DM * 2);
    __bf16* Bb    = (__bf16*)alloc((size_t)NC * BD_PAD * 2);
    float*  bnorm = (float*) alloc((size_t)NC * 4);
    __bf16* Zb    = (__bf16*)alloc((size_t)BKROWS * BD_PAD * 2);
    float*  score = (float*) alloc((size_t)BKROWS * NC * 4);
    float*  avg   = (float*) alloc((size_t)NC * 4);
    float*  loss  = (float*) alloc(256);
    int*    idx   = (int*)   alloc((size_t)BKROWS * 4);

    cvt_f32_bf16<<<(BKROWS * DM) / 256, 256, 0, stream>>>(X, Xb, BKROWS * DM);
    cvt_w_pad<<<(NPROJ_PAD * DM) / 256, 256, 0, stream>>>(W, Wb);
    cvt_basis<<<NC, 256, 0, stream>>>(basis, Bb, bnorm);
    zero_accums<<<5, 256, 0, stream>>>(avg, loss);

    // GEMM1: z[32768,2720](bf16) = X * W^T + b, pad cols zeroed
    dim3 g1(BKROWS / 64, NPROJ_PAD / 256);   // (512, 11)
    wmma_gemm_bf16<<<g1, 256, 0, stream>>>(Xb, Wb, DM, DM, DM / 32,
                                           nullptr, 0, Zb, BD_PAD, bprj, BD, BD_PAD);
    // GEMM2: score[32768,1024](f32) = Z * basis^T
    dim3 g2(BKROWS / 64, NC / 256);          // (512, 4)
    wmma_gemm_bf16<<<g2, 256, 0, stream>>>(Zb, Bb, BD_PAD, BD_PAD, BD_PAD / 32,
                                           score, NC, nullptr, 0, nullptr, NC, NC);

    vq_reduce<<<BKROWS / 8, 256, 0, stream>>>(score, bnorm, avg, idx, idx_f);
    vq_gather<<<BKROWS, 256, 0, stream>>>(basis, Zb, idx, q, loss);
    vq_final<<<1, 256, 0, stream>>>(avg, loss, scal);
}
